// Plane_refine_block_2_41927470743687
// MI455X (gfx1250) — matrix operand order repaired and checked by reading, not compiled, not run on hardware
//
#include <hip/hip_runtime.h>
#include <hip/hip_bf16.h>

// ---------------------------------------------------------------------------
// Plane_refine_block on MI455X (gfx1250, wave32):
//   Stage 0: zero pooled output + per-plane counters.
//   Stage 1: fold BN scale g into W, W1,W2 -> bf16 [128,128] row-major.
//   Stage 2: mask kernel: 1 thread/point, 64-plane mask, wave-aggregated
//            compaction into per-plane index lists (atomicAdd + ballot).
//   Stage 3: WMMA MLP: A-frags fully preloaded; B-frags group-prefetched,
//            group pinned live via empty input-only asm so the 16-load clause
//            of step ks+1 overlaps the XDL drain of step ks (one wait/group).
//   Stage 4: list-driven masked max-pool, monotone uint atomicMax (h >= 0).
// ---------------------------------------------------------------------------

typedef __attribute__((ext_vector_type(16))) __bf16        bf16x16;
typedef __attribute__((ext_vector_type(8)))  float         f32x8;
typedef __attribute__((ext_vector_type(4)))  unsigned int  u32x4;   // first-class
// NOTE: HIP's uint4 is a struct -> "indirect" inline-asm operand (unsupported);
// ext_vector u32x4 is a direct VReg_128 operand and still has .x/.y/.z/.w.

union FragAB {
    bf16x16      v;
    unsigned int u[8];
    u32x4        q[2];
};

#define PLANE_CAP 32768   // per-plane index-list capacity (expected ~4K)
#define TILE_STRIDE 136   // 128 + 8 pad: kills LDS bank conflicts on column stores

// keep all 8 B-fragments simultaneously live; single group-level waitcnt.
#define PIN_GROUP(Bg)                                                         \
    asm volatile(""                                                           \
        :: "v"((Bg)[0].q[0]), "v"((Bg)[0].q[1]),                              \
           "v"((Bg)[1].q[0]), "v"((Bg)[1].q[1]),                              \
           "v"((Bg)[2].q[0]), "v"((Bg)[2].q[1]),                              \
           "v"((Bg)[3].q[0]), "v"((Bg)[3].q[1]),                              \
           "v"((Bg)[4].q[0]), "v"((Bg)[4].q[1]),                              \
           "v"((Bg)[5].q[0]), "v"((Bg)[5].q[1]),                              \
           "v"((Bg)[6].q[0]), "v"((Bg)[6].q[1]),                              \
           "v"((Bg)[7].q[0]), "v"((Bg)[7].q[1]))

// {hi[31:16], lo[31:16]} in ONE v_perm_b32 (truncating f32->bf16 pack)
__device__ __forceinline__ unsigned int pkbf(unsigned int lo, unsigned int hi) {
    return __builtin_amdgcn_perm(hi, lo, 0x07060302u);
}
__device__ __forceinline__ unsigned short f2bf(float f) {   // RNE (prep only)
    unsigned int x = __float_as_uint(f);
    unsigned int r = x + 0x7fffu + ((x >> 16) & 1u);
    return (unsigned short)(r >> 16);
}
__device__ __forceinline__ float bf2f(unsigned short b) {
    return __uint_as_float(((unsigned int)b) << 16);
}

// B-frag: W'[jt*16+nn][ks*32 + 8*half + {0..7, 16..23}] : two 16-byte chunks
__device__ __forceinline__ void loadB(FragAB& B, const unsigned short* __restrict__ w,
                                      int ks, int jt, int nn, int half) {
    const u32x4* wp = reinterpret_cast<const u32x4*>(
        w + ((size_t)(jt * 16 + nn) * 128 + ks * 32 + 8 * half));
    B.q[0] = wp[0];
    B.q[1] = wp[2];
}

// ---------------------------------------------------------------- stage 0
__global__ void zero_kernel(float* out, int n_out, unsigned int* cnt, int n_cnt) {
    int t = blockIdx.x * blockDim.x + threadIdx.x;
    if (t < n_out) out[t] = 0.0f;
    if (t < n_cnt) cnt[t] = 0u;
}

// ---------------------------------------------------------------- stage 1
__global__ void prep_weights_kernel(const float* __restrict__ W1,
                                    const float* __restrict__ g1,
                                    const float* __restrict__ W2,
                                    const float* __restrict__ g2,
                                    unsigned short* __restrict__ w1b,
                                    unsigned short* __restrict__ w2b) {
    int t = blockIdx.x * blockDim.x + threadIdx.x;   // 0 .. 32767
    int sel = t >> 14;
    int idx = t & 16383;
    int j   = idx >> 7;
    if (sel == 0) w1b[idx] = f2bf(W1[idx] * g1[j]);
    else          w2b[idx] = f2bf(W2[idx] * g2[j]);
}

// ---------------------------------------------------------------- stage 2
__global__ __launch_bounds__(256)
void mask_kernel(const float* __restrict__ xyz, const float* __restrict__ centers,
                 const float* __restrict__ pcen, const float* __restrict__ pnrm,
                 const float* __restrict__ pmin, const float* __restrict__ pmax,
                 unsigned int* __restrict__ cnt, unsigned int* __restrict__ list,
                 int N, int P) {
    const int i = blockIdx.x * blockDim.x + threadIdx.x;
    if (i >= N) return;
    const int lane = threadIdx.x & 31;
    const unsigned int lt_mask = (1u << lane) - 1u;

    const float x = xyz[3 * i + 0] + centers[0];
    const float y = xyz[3 * i + 1] + centers[1];
    const float z = xyz[3 * i + 2] + centers[2];

    for (int p = 0; p < P; ++p) {               // plane params uniform -> s_loads
        const float nx = pnrm[p * 3 + 0], ny = pnrm[p * 3 + 1], nz = pnrm[p * 3 + 2];
        const float off = pcen[p * 3 + 0] * nx + pcen[p * 3 + 1] * ny
                        + pcen[p * 3 + 2] * nz;
        const float mn0 = pmin[p * 3 + 0], mn1 = pmin[p * 3 + 1], mn2 = pmin[p * 3 + 2];
        const float mx0 = pmax[p * 3 + 0], mx1 = pmax[p * 3 + 1], mx2 = pmax[p * 3 + 2];
        bool ok = fabsf(x * nx + y * ny + z * nz - off) < 0.05f;
        ok = ok && ((mx0 == 0.0f) || (x >= mn0 && x < mx0));
        ok = ok && ((mx1 == 0.0f) || (y >= mn1 && y < mx1));
        ok = ok && ((mx2 == 0.0f) || (z >= mn2 && z < mx2));

        unsigned int bal = __builtin_amdgcn_ballot_w32(ok);
        unsigned int nset = __popc(bal);
        unsigned int base0 = 0u;
        if (lane == 0 && nset) base0 = atomicAdd(&cnt[p], nset);
        unsigned int base = __builtin_amdgcn_readfirstlane(base0);
        if (ok) {
            unsigned int pos = base + __popc(bal & lt_mask);
            if (pos < PLANE_CAP) list[(size_t)p * PLANE_CAP + pos] = (unsigned int)i;
        }
    }
}

// ---------------------------------------------------------------- stage 3
__global__ __launch_bounds__(256)
void mlp_wmma_kernel(const float* __restrict__ feat,           // [N,128] f32
                     const unsigned short* __restrict__ w1b,   // [128,128] bf16
                     const unsigned short* __restrict__ w2b,   // [128,128] bf16
                     const float* __restrict__ b1,
                     const float* __restrict__ b2,
                     unsigned short* __restrict__ hout,        // [N,128] bf16
                     int N) {
    __shared__ unsigned short tile[8][16 * TILE_STRIDE];       // 34 KB

    const int lane = threadIdx.x & 31;
    const int wave = threadIdx.x >> 5;
    const int nn   = lane & 15;
    const int half = lane >> 4;

    const int base = (blockIdx.x * 8 + wave) * 16;
    if (base >= N) return;
    unsigned short* my = tile[wave];

    int rowA = base + nn; if (rowA > N - 1) rowA = N - 1;

    // ---- preload + pack ALL layer-1 A-fragments (one big load clause) -----
    u32x4 raw[4][4];
    #pragma unroll
    for (int ks = 0; ks < 4; ++ks) {
        const u32x4* rp = reinterpret_cast<const u32x4*>(
            feat + (size_t)rowA * 128 + ks * 32 + 8 * half);
        raw[ks][0] = rp[0]; raw[ks][1] = rp[1];   // K +0..7   (f32 bits)
        raw[ks][2] = rp[4]; raw[ks][3] = rp[5];   // K +16..23
    }
    FragAB A1[4];
    #pragma unroll
    for (int ks = 0; ks < 4; ++ks) {
        A1[ks].u[0] = pkbf(raw[ks][0].x, raw[ks][0].y);
        A1[ks].u[1] = pkbf(raw[ks][0].z, raw[ks][0].w);
        A1[ks].u[2] = pkbf(raw[ks][1].x, raw[ks][1].y);
        A1[ks].u[3] = pkbf(raw[ks][1].z, raw[ks][1].w);
        A1[ks].u[4] = pkbf(raw[ks][2].x, raw[ks][2].y);
        A1[ks].u[5] = pkbf(raw[ks][2].z, raw[ks][2].w);
        A1[ks].u[6] = pkbf(raw[ks][3].x, raw[ks][3].y);
        A1[ks].u[7] = pkbf(raw[ks][3].z, raw[ks][3].w);
    }

    // ---- layer 1: acc = b1 (bias as WMMA C-init) --------------------------
    f32x8 acc1[8];
    #pragma unroll
    for (int jt = 0; jt < 8; ++jt) {
        const float b = b1[jt * 16 + nn];
        #pragma unroll
        for (int r = 0; r < 8; ++r) acc1[jt][r] = b;
    }

    {
        FragAB Bg[8];
        #pragma unroll
        for (int ks = 0; ks < 4; ++ks) {
            #pragma unroll
            for (int jt = 0; jt < 8; ++jt) loadB(Bg[jt], w1b, ks, jt, nn, half);
            PIN_GROUP(Bg);   // force 8 live frags + one group-level wait
            #pragma unroll
            for (int jt = 0; jt < 8; ++jt)
                acc1[jt] = __builtin_amdgcn_wmma_f32_16x16x32_bf16(
                    false, A1[ks].v, false, Bg[jt].v, (short)0, acc1[jt], false, false);
        }
    }

    // relu + truncating bf16 store (ds_store_b16_d16_hi: data[31:16])
    #pragma unroll
    for (int jt = 0; jt < 8; ++jt) {
        #pragma unroll
        for (int r = 0; r < 8; ++r) {
            float v = fmaxf(acc1[jt][r], 0.0f);
            my[(8 * half + r) * TILE_STRIDE + jt * 16 + nn] =
                (unsigned short)(__float_as_uint(v) >> 16);
        }
    }
    __asm__ volatile("s_wait_dscnt 0" ::: "memory");   // wave-private tile ready

    // ---- layer 2: A-frags straight from the bf16 tile ---------------------
    FragAB A2[4];
    #pragma unroll
    for (int ks = 0; ks < 4; ++ks) {
        const u32x4* lp = reinterpret_cast<const u32x4*>(
            &my[nn * TILE_STRIDE + ks * 32 + 8 * half]);
        A2[ks].q[0] = lp[0];
        A2[ks].q[1] = lp[2];                     // +16 bf16 = +2 u32x4
    }

    f32x8 acc2[8];
    #pragma unroll
    for (int jt = 0; jt < 8; ++jt) {
        const float b = b2[jt * 16 + nn];
        #pragma unroll
        for (int r = 0; r < 8; ++r) acc2[jt][r] = b;
    }

    {
        FragAB Bg[8];
        #pragma unroll
        for (int ks = 0; ks < 4; ++ks) {
            #pragma unroll
            for (int jt = 0; jt < 8; ++jt) loadB(Bg[jt], w2b, ks, jt, nn, half);
            PIN_GROUP(Bg);
            #pragma unroll
            for (int jt = 0; jt < 8; ++jt)
                acc2[jt] = __builtin_amdgcn_wmma_f32_16x16x32_bf16(
                    false, A2[ks].v, false, Bg[jt].v, (short)0, acc2[jt], false, false);
        }
    }

    // relu + restage (LDS ops in-order per wave: can't pass the A loads above)
    #pragma unroll
    for (int jt = 0; jt < 8; ++jt) {
        #pragma unroll
        for (int r = 0; r < 8; ++r) {
            float v = fmaxf(acc2[jt][r], 0.0f);
            my[(8 * half + r) * TILE_STRIDE + jt * 16 + nn] =
                (unsigned short)(__float_as_uint(v) >> 16);
        }
    }
    __asm__ volatile("s_wait_dscnt 0" ::: "memory");

    // coalesced tile -> global: 256 chunks of 16B, 8 per lane
    #pragma unroll
    for (int t = 0; t < 8; ++t) {
        int cid  = t * 32 + lane;
        int rrow = cid >> 4;
        int seg  = cid & 15;
        if (base + rrow < N) {
            const u32x4* s = reinterpret_cast<const u32x4*>(
                &my[rrow * TILE_STRIDE + seg * 8]);
            reinterpret_cast<u32x4*>(hout + (size_t)(base + rrow) * 128 + seg * 8)[0] = s[0];
        }
    }
}

// ---------------------------------------------------------------- stage 4
__global__ __launch_bounds__(128)
void pool_kernel(const unsigned short* __restrict__ h,       // [N,128] bf16
                 const unsigned int* __restrict__ cnt,
                 const unsigned int* __restrict__ list,
                 float* __restrict__ out,                    // [P,128]
                 int P, int S) {
    const int p = blockIdx.x % P;
    const int s = blockIdx.x / P;
    const int d = threadIdx.x;

    unsigned int n = cnt[p];
    if (n > PLANE_CAP) n = PLANE_CAP;
    const unsigned int seg = (n + S - 1) / S;
    const unsigned int j0  = s * seg;
    unsigned int j1 = j0 + seg; if (j1 > n) j1 = n;

    float m = 0.0f;                      // relu => h >= 0; empty mask -> 0
    for (unsigned int j = j0; j < j1; ++j) {
        const unsigned int idx = list[(size_t)p * PLANE_CAP + j];   // uniform
        m = fmaxf(m, bf2f(h[(size_t)idx * 128 + d]));               // 256B row, coalesced
    }
    atomicMax(reinterpret_cast<unsigned int*>(&out[p * 128 + d]), __float_as_uint(m));
}

// ---------------------------------------------------------------------------
extern "C" void kernel_launch(void* const* d_in, const int* in_sizes, int n_in,
                              void* d_out, int out_size, void* d_ws, size_t ws_size,
                              hipStream_t stream) {
    const float* feature = (const float*)d_in[0];   // [N,128]
    const float* xyz     = (const float*)d_in[1];   // [N,3]
    const float* centers = (const float*)d_in[2];   // [3]
    const float* pcen    = (const float*)d_in[3];   // [P,3]
    const float* pnrm    = (const float*)d_in[4];
    const float* pmin    = (const float*)d_in[5];
    const float* pmax    = (const float*)d_in[6];
    const float* W1      = (const float*)d_in[7];
    const float* g1      = (const float*)d_in[8];
    const float* b1      = (const float*)d_in[9];
    const float* W2      = (const float*)d_in[10];
    const float* g2      = (const float*)d_in[11];
    const float* b2      = (const float*)d_in[12];
    float* out = (float*)d_out;                     // [P,128]

    const int N = in_sizes[0] / 128;
    const int P = in_sizes[3] / 3;

    // workspace layout
    unsigned short* w1b  = (unsigned short*)d_ws;                 // 16384
    unsigned short* w2b  = w1b + 16384;                           // 16384
    unsigned short* hbuf = w2b + 16384;                           // N*128
    unsigned int*   cnt  = (unsigned int*)(hbuf + (size_t)N * 128);      // P
    unsigned int*   list = cnt + ((P + 63) & ~63);                // P*PLANE_CAP
    (void)ws_size;  // needs ~64KB + N*256B + P*4 + P*PLANE_CAP*4  (~40MB)

    zero_kernel<<<(P * 128 + 255) / 256, 256, 0, stream>>>(out, P * 128, cnt, P);

    prep_weights_kernel<<<128, 256, 0, stream>>>(W1, g1, W2, g2, w1b, w2b);

    mask_kernel<<<(N + 255) / 256, 256, 0, stream>>>(xyz, centers, pcen, pnrm,
                                                     pmin, pmax, cnt, list, N, P);

    const int nblk = (N + 127) / 128;   // 8 waves * 16 pts / block
    mlp_wmma_kernel<<<nblk, 256, 0, stream>>>(feature, w1b, w2b, b1, b2, hbuf, N);

    const int S = 16;                   // splits per plane
    pool_kernel<<<P * S, 128, 0, stream>>>(hbuf, cnt, list, out, P, S);
}
